// RNNBasedEBM_4535485464801
// MI455X (gfx1250) — matrix operand order repaired and checked by reading, not compiled, not run on hardware
//
#include <hip/hip_runtime.h>

#define BATCH  256
#define TSTEPS 512
#define IDIM   256
#define HDIM   512
#define GDIM   1536   // 3*HDIM

#define HH_PACK_BYTES (GDIM * HDIM * 2)          // 1,572,864 : 96 gt-tiles x 16 kt x 1KB
#define IH_PACK_BYTES (GDIM * IDIM * 2)          //   786,432 : 96 gt-tiles x  8 kt x 1KB
#define XBF_OFF       ((size_t)(HH_PACK_BYTES + IH_PACK_BYTES))
#define XBF_BYTES     ((size_t)BATCH * TSTEPS * IDIM * 2)  // 67 MB

#if __has_builtin(__builtin_amdgcn_sched_group_barrier)
#define SGB(mask, n) __builtin_amdgcn_sched_group_barrier((mask), (n), 0)
#else
#define SGB(mask, n)
#endif
// masks: 0x008 = MFMA/WMMA, 0x020 = VMEM read, 0x100 = DS read

typedef __attribute__((ext_vector_type(16))) __bf16 bf16x16;
typedef __attribute__((ext_vector_type(8)))  float  f32x8;

__device__ __forceinline__ unsigned short f32_to_bf16_bits(float f) {
  unsigned u = __builtin_bit_cast(unsigned, f);
  u += 0x7FFFu + ((u >> 16) & 1u);  // RNE
  return (unsigned short)(u >> 16);
}

__device__ __forceinline__ f32x8 wmma_bf16(bf16x16 a, bf16x16 b, f32x8 c) {
  return __builtin_amdgcn_wmma_f32_16x16x32_bf16(false, a, false, b, (short)0, c,
                                                 false, false);
}

// Byte offset of element (row m, col kp) inside a 16x32 bf16 A-tile stored in
// WMMA fragment order (32 B per lane, 1 KB per tile).
__device__ __forceinline__ int afrag_byte(int m, int kp) {
  int lane = m + ((kp & 8) ? 16 : 0);
  int v    = ((kp & 7) >> 1) + ((kp >> 4) & 1) * 4;
  return lane * 32 + v * 4 + (kp & 1) * 2;
}

// ---------------------------------------------------------------------------
// Pack weights f32 -> bf16 in B-FRAGMENT-MAJOR order:
//   block(gt, kt) = 1 KB, lane-contiguous: lane&15 = N col, lane>>4 = K half.
// ---------------------------------------------------------------------------
__global__ void pack_weights_kernel(const float* __restrict__ w_ih,
                                    const float* __restrict__ w_hh,
                                    unsigned short* __restrict__ wpk) {
  int o = blockIdx.x * blockDim.x + threadIdx.x;
  const int nhh = GDIM * HDIM;
  const int nih = GDIM * IDIM;
  if (o < nhh) {
    int blk = o >> 9, r = o & 511;
    int lane = r >> 4, e = r & 15;
    int gt = blk >> 4, kt = blk & 15;           // 96 x 16
    int g = gt * 16 + (lane & 15);
    int k = kt * 32 + (lane >> 4) * 8 + (e & 7) + (e >> 3) * 16;
    wpk[o] = f32_to_bf16_bits(w_hh[g * HDIM + k]);
  } else if (o < nhh + nih) {
    int oo = o - nhh;
    int blk = oo >> 9, r = oo & 511;
    int lane = r >> 4, e = r & 15;
    int gt = blk >> 3, kt = blk & 7;            // 96 x 8
    int g = gt * 16 + (lane & 15);
    int k = kt * 32 + (lane >> 4) * 8 + (e & 7) + (e >> 3) * 16;
    wpk[nhh + oo] = f32_to_bf16_bits(w_ih[g * IDIM + k]);
  }
}

// One-time x f32 -> bf16 (HBM-bound, ~6 us at 23.3 TB/s). Optional (ws-size gated).
__global__ void convert_x_kernel(const float* __restrict__ xf,
                                 unsigned short* __restrict__ xb) {
  size_t i4 = ((size_t)blockIdx.x * blockDim.x + threadIdx.x) * 4;
  float4 f = *(const float4*)(xf + i4);
  unsigned v0 = (unsigned)f32_to_bf16_bits(f.x) | ((unsigned)f32_to_bf16_bits(f.y) << 16);
  unsigned v1 = (unsigned)f32_to_bf16_bits(f.z) | ((unsigned)f32_to_bf16_bits(f.w) << 16);
  uint2 v; v.x = v0; v.y = v1;
  *(uint2*)(xb + i4) = v;
}

// ---------------------------------------------------------------------------
// Persistent GRU: 16 WGs (one 16-row batch tile each), 512 thr = 16 waves.
// Wave owns 32 hidden cols -> 6 gate N-tiles. h kept in REGISTERS (f32) +
// bf16 A-fragment LDS copy for the next step's GEMM. Weights stream from L2.
// sched_group_barrier pins a 1-iteration load->WMMA software pipeline.
// ---------------------------------------------------------------------------
__global__ __launch_bounds__(512, 1) void gru_ebm_kernel(
    const float* __restrict__ x,
    const unsigned short* __restrict__ xbf,   // nullptr -> convert in-kernel
    const float* __restrict__ b,
    const float* __restrict__ b_n,
    const float* __restrict__ w_proj,
    const float* __restrict__ b_proj,
    const unsigned short* __restrict__ wpk,
    float* __restrict__ out) {
  __shared__ __align__(128) unsigned char sh_hfrag[16 * 1024];     // 16 K-tiles
  __shared__ __align__(128) unsigned char sh_xfrag[2 * 8 * 1024];  // double-buffered
  __shared__ float sh_red[16][17];

  const int tid  = threadIdx.x;
  const int w    = tid >> 5;
  const int lane = tid & 31;
  const int b0   = blockIdx.x * 16;

  for (int i = tid; i < 16 * 1024 / 4; i += 512) ((unsigned*)sh_hfrag)[i] = 0u;

  const int ncol  = lane & 15;
  const int mbase = (lane & 16) ? 8 : 0;
  const int cb    = w * 32;

  // packed-block byte offsets for this wave's 6 gate tiles (j={r0,r1,z0,z1,n0,n1})
  int off_hh[6], off_ih[6];
#pragma unroll
  for (int j = 0; j < 6; ++j) {
    int gt = (j >> 1) * 32 + w * 2 + (j & 1);
    off_hh[j] = gt * 16 * 1024 + lane * 32;
    off_ih[j] = HH_PACK_BYTES + gt * 8 * 1024 + lane * 32;
  }

  float bR[2], bZ[2], bN[2], bNn[2];
#pragma unroll
  for (int s = 0; s < 2; ++s) {
    int c  = cb + s * 16 + ncol;
    bR[s]  = b[c];
    bZ[s]  = b[HDIM + c];
    bN[s]  = b[2 * HDIM + c];
    bNn[s] = b_n[c];
  }

  float h_[2][8];
#pragma unroll
  for (int s = 0; s < 2; ++s)
#pragma unroll
    for (int i = 0; i < 8; ++i) h_[s][i] = 0.f;

  // x staging: thread = one 16-byte chunk (row m, 8 consecutive k) -> one
  // aligned 16B LDS store into fragment order.
  const int sm   = tid >> 5;
  const int k8   = tid & 31;
  const int sdst = (k8 >> 2) * 1024 + (sm + ((k8 & 1) << 4)) * 32 + ((k8 >> 1) & 1) * 16;
  const size_t srow = ((size_t)(b0 + sm)) * TSTEPS;

  auto stage_x = [&](int t, unsigned char* dstbuf) {
    size_t so = (srow + t) * IDIM + k8 * 8;
    if (xbf) {
      int4 v = *(const int4*)(xbf + so);
      *(int4*)(dstbuf + sdst) = v;
    } else {
      float4 f0 = *(const float4*)(x + so);
      float4 f1 = *(const float4*)(x + so + 4);
      int4 v;
      v.x = (int)((unsigned)f32_to_bf16_bits(f0.x) | ((unsigned)f32_to_bf16_bits(f0.y) << 16));
      v.y = (int)((unsigned)f32_to_bf16_bits(f0.z) | ((unsigned)f32_to_bf16_bits(f0.w) << 16));
      v.z = (int)((unsigned)f32_to_bf16_bits(f1.x) | ((unsigned)f32_to_bf16_bits(f1.y) << 16));
      v.w = (int)((unsigned)f32_to_bf16_bits(f1.z) | ((unsigned)f32_to_bf16_bits(f1.w) << 16));
      *(int4*)(dstbuf + sdst) = v;
    }
  };

  stage_x(0, sh_xfrag);
  __syncthreads();

  int p = 0;
  for (int t = 0; t < TSTEPS; ++t) {
    // prefetch next step's x into the other buffer (overlaps with GEMM below)
    if (t + 1 < TSTEPS) stage_x(t + 1, sh_xfrag + (p ^ 1) * 8192);

    // Launder a 32-bit ZERO offset each step: keeps global address-space
    // provenance (global_load saddr form) but blocks hoisting the t-invariant
    // weight loads out of the T-loop.
    unsigned zoff = 0;
    asm volatile("" : "+s"(zoff));
    const unsigned char* wb = (const unsigned char*)wpk + zoff;
    const unsigned char* xb = sh_xfrag + p * 8192;

    f32x8 aR0{}, aR1{}, aZ0{}, aZ1{}, aIN0{}, aIN1{}, aHN0{}, aHN1{};

    // ---- input-gate GEMM: (16 x 256) @ (256 x 3H) slice, pipelined ----
    {
      bf16x16 a_c = *(const bf16x16*)(xb + lane * 32);
      bf16x16 bc[6];
#pragma unroll
      for (int j = 0; j < 6; ++j) bc[j] = *(const bf16x16*)(wb + off_ih[j]);
#pragma unroll
      for (int kt = 0; kt < 8; ++kt) {
        int ktn = (kt + 1 < 8) ? kt + 1 : kt;
        bf16x16 a_n = *(const bf16x16*)(xb + ktn * 1024 + lane * 32);
        bf16x16 bn[6];
#pragma unroll
        for (int j = 0; j < 6; ++j)
          bn[j] = *(const bf16x16*)(wb + off_ih[j] + ktn * 1024);
        aR0  = wmma_bf16(a_c, bc[0], aR0);
        aR1  = wmma_bf16(a_c, bc[1], aR1);
        aZ0  = wmma_bf16(a_c, bc[2], aZ0);
        aZ1  = wmma_bf16(a_c, bc[3], aZ1);
        aIN0 = wmma_bf16(a_c, bc[4], aIN0);
        aIN1 = wmma_bf16(a_c, bc[5], aIN1);
        // pin pipeline: next-kt loads issue before this kt's WMMAs retire
        SGB(0x020, 12);  // 12 VMEM reads  (6 B-fragments)
        SGB(0x100, 2);   //  2 DS reads    (next A-fragment)
        SGB(0x008, 6);   //  6 WMMA
        a_c = a_n;
#pragma unroll
        for (int j = 0; j < 6; ++j) bc[j] = bn[j];
      }
    }
    // ---- hidden GEMM: (16 x 512) @ (512 x 3H) slice, pipelined ----
    {
      bf16x16 a_c = *(const bf16x16*)(sh_hfrag + lane * 32);
      bf16x16 bc[6];
#pragma unroll
      for (int j = 0; j < 6; ++j) bc[j] = *(const bf16x16*)(wb + off_hh[j]);
#pragma unroll
      for (int kt = 0; kt < 16; ++kt) {
        int ktn = (kt + 1 < 16) ? kt + 1 : kt;
        bf16x16 a_n = *(const bf16x16*)(sh_hfrag + ktn * 1024 + lane * 32);
        bf16x16 bn[6];
#pragma unroll
        for (int j = 0; j < 6; ++j)
          bn[j] = *(const bf16x16*)(wb + off_hh[j] + ktn * 1024);
        aR0  = wmma_bf16(a_c, bc[0], aR0);
        aR1  = wmma_bf16(a_c, bc[1], aR1);
        aZ0  = wmma_bf16(a_c, bc[2], aZ0);
        aZ1  = wmma_bf16(a_c, bc[3], aZ1);
        aHN0 = wmma_bf16(a_c, bc[4], aHN0);
        aHN1 = wmma_bf16(a_c, bc[5], aHN1);
        SGB(0x020, 12);
        SGB(0x100, 2);
        SGB(0x008, 6);
        a_c = a_n;
#pragma unroll
        for (int j = 0; j < 6; ++j) bc[j] = bn[j];
      }
    }
    __syncthreads();   // all hfrag reads done before update rewrites it

    // ---- register-resident GRU update for this wave's 32 hidden columns ----
#pragma unroll
    for (int s = 0; s < 2; ++s) {
      const f32x8 accR  = s ? aR1  : aR0;
      const f32x8 accZ  = s ? aZ1  : aZ0;
      const f32x8 accIN = s ? aIN1 : aIN0;
      const f32x8 accHN = s ? aHN1 : aHN0;
      int c = cb + s * 16 + ncol;
#pragma unroll
      for (int i = 0; i < 8; ++i) {
        float rv = 1.f / (1.f + __expf(-(accR[i] + bR[s])));
        float zv = 1.f / (1.f + __expf(-(accZ[i] + bZ[s])));
        float q  = accIN[i] + bN[s] + rv * (accHN[i] + bNn[s]);
        float e2 = __expf(-2.f * fabsf(q));
        float nv = copysignf((1.f - e2) / (1.f + e2), q);   // tanh(q)
        float hn = nv + zv * (h_[s][i] - nv);               // n + z*(h - n)
        h_[s][i] = hn;
        *(unsigned short*)(sh_hfrag + (c >> 5) * 1024 + afrag_byte(mbase + i, c & 31)) =
            f32_to_bf16_bits(hn);
      }
    }
    __syncthreads();   // hfrag update visible before next step's GEMM
    p ^= 1;
  }

  // ---- final projection from registers: out = h . w_proj + b_proj ----
  float wp0 = w_proj[cb + ncol];
  float wp1 = w_proj[cb + 16 + ncol];
#pragma unroll
  for (int i = 0; i < 8; ++i) {
    float acc = h_[0][i] * wp0 + h_[1][i] * wp1;
#pragma unroll
    for (int d = 1; d <= 8; d <<= 1) acc += __shfl_xor(acc, d, 32);
    if (ncol == 0) sh_red[mbase + i][w] = acc;  // lanes 0 and 16 write
  }
  __syncthreads();
  if (tid < 16) {
    float s = 0.f;
#pragma unroll
    for (int wv = 0; wv < 16; ++wv) s += sh_red[tid][wv];
    out[b0 + tid] = s + b_proj[0];
  }
}

// ---------------------------------------------------------------------------
extern "C" void kernel_launch(void* const* d_in, const int* in_sizes, int n_in,
                              void* d_out, int out_size, void* d_ws, size_t ws_size,
                              hipStream_t stream) {
  const float* x      = (const float*)d_in[0];
  const float* w_ih   = (const float*)d_in[1];
  const float* w_hh   = (const float*)d_in[2];
  const float* b      = (const float*)d_in[3];
  const float* b_n    = (const float*)d_in[4];
  const float* w_proj = (const float*)d_in[5];
  const float* b_proj = (const float*)d_in[6];
  float* out = (float*)d_out;
  unsigned short* wpk = (unsigned short*)d_ws;

  int nconv = GDIM * (HDIM + IDIM);
  pack_weights_kernel<<<(nconv + 255) / 256, 256, 0, stream>>>(w_ih, w_hh, wpk);

  const bool x_pre = ws_size >= XBF_OFF + XBF_BYTES;
  unsigned short* xbf = nullptr;
  if (x_pre) {
    xbf = (unsigned short*)((char*)d_ws + XBF_OFF);
    size_t n4 = ((size_t)BATCH * TSTEPS * IDIM) / 4;   // 8,388,608
    convert_x_kernel<<<(unsigned)(n4 / 256), 256, 0, stream>>>(x, xbf);
  }

  gru_ebm_kernel<<<BATCH / 16, 512, 0, stream>>>(x, xbf, b, b_n, w_proj, b_proj,
                                                 wpk, out);
}
